// DAttention_v2_76579266887927
// MI455X (gfx1250) — matrix operand order repaired and checked by reading, not compiled, not run on hardware
//
#include <hip/hip_runtime.h>
#include <math.h>

typedef __attribute__((ext_vector_type(16))) _Float16 v16h;
typedef __attribute__((ext_vector_type(8)))  float    v8f;

#define K_REFN 4
#define K_RN   2
#define K_NH   4
#define K_DH   32
#define K_C    128
#define K_H    64
#define K_W    64
#define K_HK   32
#define K_WK   32
#define K_NS   1024
#define K_L    2048
#define K_B    8
#define K_SCALE 0.17677669529663687f   /* 32^-0.5 */
#define K_OFR   2.0f

union AFrag { v16h v; _Float16 h[16]; };
union CFrag { v8f  v; float    f[8];  };

static __device__ __forceinline__ v8f wmma16(v16h a, v16h b, v8f c) {
  // D = A(16x32 f16) * B(32x16 f16) + C(16x16 f32)
  return __builtin_amdgcn_wmma_f32_16x16x32_f16(
      /*neg_a=*/false, a, /*neg_b=*/false, b,
      /*c_mod=*/(short)0, c, /*reuse_a=*/false, /*reuse_b=*/false);
}

// ---------------------------------------------------------------------------
// Batched WMMA GEMM, K fixed at 128, all variants compile-time specialized:
//   D[bz] = A[bz] * B[bz % bmodB] (+biasM per row / +biasN per col)
// A row-major [M,128]; B row-major [128,N] (TRANSB=0) or [N,128] (TRANSB=1).
// One wave per 16x16 output tile.  grid = (N/16, M/16, batch), block = 32.
// ---------------------------------------------------------------------------
template <int TRANSB, int HAS_BM, int HAS_BN>
__global__ void gemm_wmma_kernel(const float* __restrict__ A,
                                 const float* __restrict__ Bm,
                                 const float* __restrict__ biasM,
                                 const float* __restrict__ biasN,
                                 float* __restrict__ D,
                                 int lda, int ldb, int ldd,
                                 long sA, long sB, long sD,
                                 int bmodB)
{
  const int lane = threadIdx.x & 31;
  const int lm = lane & 15;          // A: row-in-tile, B/D: col-in-tile
  const int kh = lane >> 4;          // K-half selector
  const int nt = blockIdx.x, mt = blockIdx.y, bz = blockIdx.z;

  const float* Ab = A  + (long)bz * sA;
  const float* Bb = Bm + (long)(bz % bmodB) * sB;
  float*       Db = D  + (long)bz * sD;

  const float* arow = Ab + (long)(mt * 16 + lm) * lda;
  // TRANSB: fixed row per lane; else column pointer
  const float* bbase = TRANSB ? (Bb + (long)(nt * 16 + lm) * ldb)
                              : (Bb + nt * 16 + lm);

  CFrag acc;
  #pragma unroll
  for (int r = 0; r < 8; ++r) acc.f[r] = 0.0f;

  #pragma unroll
  for (int k0 = 0; k0 < K_C; k0 += 32) {
    AFrag a, b;
    #pragma unroll
    for (int j = 0; j < 16; ++j) {
      // A 16x32 f16 layout: lane(0-15)=M, elem j -> K = {0..7 | 16..23} + 8*kh
      int kk = k0 + ((j < 8) ? 0 : 16) + kh * 8 + (j & 7);
      a.h[j] = (_Float16)arow[kk];
    }
    #pragma unroll
    for (int j = 0; j < 16; ++j) {
      // B 32x16 f16 layout: lane(0-15)=N, elem j -> K = {0..7 | 16..23} + 8*kh
      int kk = k0 + ((j < 8) ? 0 : 16) + kh * 8 + (j & 7);
      float bv = TRANSB ? bbase[kk] : bbase[(long)kk * ldb];
      b.h[j] = (_Float16)bv;
    }
    acc.v = wmma16(a.v, b.v, acc.v);
  }

  #pragma unroll
  for (int r = 0; r < 8; ++r) {
    // C/D layout: elem r -> M = r + 8*kh, N = lm
    int m = mt * 16 + r + 8 * kh;
    int n = nt * 16 + lm;
    float v = acc.f[r];
    if (HAS_BM) v += biasM[m];
    if (HAS_BN) v += biasN[n];
    Db[(long)m * ldd + n] = v;
  }
}

// ---------------------------------------------------------------------------
// Stage 1: masked mean of q_ over L  ->  qm[b,c]
// ---------------------------------------------------------------------------
__global__ void qmean_kernel(const float* __restrict__ q_,
                             const float* __restrict__ mask,
                             float* __restrict__ qm)
{
  int b = blockIdx.x, c = threadIdx.x;
  float s = 0.0f, ms = 0.0f;
  for (int l = 0; l < K_L; ++l) {
    float mk = mask[(long)b * K_L + l];
    s  += q_[((long)b * K_L + l) * K_C + c] * mk;
    ms += mk;
  }
  qm[b * K_C + c] = s / (ms + 1e-6f);
}

// Stage 2: q_cond = qm @ Wq^T + bq
__global__ void qcond_kernel(const float* __restrict__ qm,
                             const float* __restrict__ Wq,
                             const float* __restrict__ bq,
                             float* __restrict__ qcond)
{
  int b = blockIdx.x, o = threadIdx.x;
  float s = bq[o];
  for (int i = 0; i < K_C; ++i) s += qm[b * K_C + i] * Wq[o * K_C + i];
  qcond[b * K_C + o] = s;
}

// Stage 3: wmod[b,o,i] = normalize_i( Wmd[o,i] * (q_cond[b,i]+1) )
__global__ void wmod_kernel(const float* __restrict__ Wmd,
                            const float* __restrict__ qcond,
                            float* __restrict__ wmod)
{
  int bo = blockIdx.x;
  int b = bo >> 7, o = bo & 127;
  int i = threadIdx.x;
  float w = Wmd[o * K_C + i] * (qcond[b * K_C + i] + 1.0f);
  __shared__ float red[K_C];
  red[i] = w * w;
  __syncthreads();
  for (int st = 64; st > 0; st >>= 1) {
    if (i < st) red[i] += red[i + st];
    __syncthreads();
  }
  float scale = rsqrtf(red[0] + 1e-8f);
  wmod[((long)b * K_C + o) * K_C + i] = w * scale;
}

// Stage 5: depthwise 3x3 stride-2 pad-1 conv, q_t[8,128,64,64] -> t0[8,128,32,32]
__global__ void dwconv_kernel(const float* __restrict__ q_t,
                              const float* __restrict__ dw_w,
                              const float* __restrict__ dw_b,
                              float* __restrict__ t0)
{
  int idx = blockIdx.x * blockDim.x + threadIdx.x;     // B*C*NS
  int xo = idx & 31, yo = (idx >> 5) & 31;
  int c = (idx >> 10) & 127, b = idx >> 17;
  float s = dw_b[c];
  const float* in = q_t + ((long)b * K_C + c) * (K_H * K_W);
  #pragma unroll
  for (int ky = 0; ky < 3; ++ky) {
    int yi = yo * 2 - 1 + ky;
    if (yi < 0 || yi >= K_H) continue;
    #pragma unroll
    for (int kx = 0; kx < 3; ++kx) {
      int xi = xo * 2 - 1 + kx;
      if (xi < 0 || xi >= K_W) continue;
      s += in[yi * K_W + xi] * dw_w[c * 9 + ky * 3 + kx];
    }
  }
  t0[((long)b * K_C + c) * K_NS + yo * K_WK + xo] = s;
}

// Stage 6: channel LayerNorm + exact GELU (in place on t0)
__global__ void lngelu_kernel(float* __restrict__ t0,
                              const float* __restrict__ ln_g,
                              const float* __restrict__ ln_b)
{
  int p = blockIdx.x;                // B*NS positions
  int b = p >> 10, sp = p & (K_NS - 1);
  int c = threadIdx.x;
  float v = t0[((long)b * K_C + c) * K_NS + sp];
  __shared__ float r1[K_C], r2[K_C];
  r1[c] = v; r2[c] = v * v;
  __syncthreads();
  for (int st = 64; st > 0; st >>= 1) {
    if (c < st) { r1[c] += r1[c + st]; r2[c] += r2[c + st]; }
    __syncthreads();
  }
  float mean = r1[0] * (1.0f / K_C);
  float var  = r2[0] * (1.0f / K_C) - mean * mean;
  float xn = (v - mean) * rsqrtf(var + 1e-5f) * ln_g[c] + ln_b[c];
  float ge = 0.5f * xn * (1.0f + erff(xn * 0.70710678118654752f));
  t0[((long)b * K_C + c) * K_NS + sp] = ge;
}

// Stage 7: offset head -> pos (gx,gy), plus grid/ref outputs
__global__ void offset_kernel(const float* __restrict__ t0,
                              const float* __restrict__ off_w,
                              float* __restrict__ pos,
                              float* __restrict__ gout,
                              float* __restrict__ rout)
{
  int idx = blockIdx.x * blockDim.x + threadIdx.x;   // B*NS
  int sp = idx & (K_NS - 1), b = idx >> 10;
  int hk = sp >> 5, wk = sp & 31;
  float s0 = 0.0f, s1 = 0.0f;
  const float* tb = t0 + (long)b * K_C * K_NS + sp;
  for (int c = 0; c < K_C; ++c) {
    float tv = tb[(long)c * K_NS];
    s0 += off_w[c] * tv;
    s1 += off_w[K_C + c] * tv;
  }
  float offy = tanhf(s0) * (K_OFR / (float)(K_HK - 1));
  float offx = tanhf(s1) * (K_OFR / (float)(K_WK - 1));
  float ry = ((float)hk + 0.5f) * (2.0f / (float)(K_HK - 1)) - 1.0f;
  float rx = ((float)wk + 0.5f) * (2.0f / (float)(K_WK - 1)) - 1.0f;
  float py = offy + ry, px = offx + rx;       // pos[...,::-1] -> (x,y)
  pos[((long)b * K_NS + sp) * 2 + 0] = px;
  pos[((long)b * K_NS + sp) * 2 + 1] = py;
  // grid output batch order: b = rn*REFN + refn  ->  b' = refn*RN + rn
  int rn = b >> 2, refn = b & 3;
  int bo = refn * K_RN + rn;
  gout[((long)bo * K_NS + sp) * 2 + 0] = px;
  gout[((long)bo * K_NS + sp) * 2 + 1] = py;
  rout[((long)b * K_NS + sp) * 2 + 0] = ry;   // ref stays (y,x)
  rout[((long)b * K_NS + sp) * 2 + 1] = rx;
}

// Stage 8: bilinear grid sample of y -> xs[8,128,1024]
__global__ void gsample_kernel(const float* __restrict__ yimg,
                               const float* __restrict__ pos,
                               float* __restrict__ xs)
{
  int idx = blockIdx.x * blockDim.x + threadIdx.x;   // B*C*NS
  int n = idx & (K_NS - 1);
  int c = (idx >> 10) & (K_C - 1);
  int b = idx >> 17;
  float gx = pos[((long)b * K_NS + n) * 2 + 0];
  float gy = pos[((long)b * K_NS + n) * 2 + 1];
  float fx = (gx + 1.0f) * 0.5f * (K_W - 1);
  float fy = (gy + 1.0f) * 0.5f * (K_H - 1);
  float x0 = floorf(fx), y0 = floorf(fy);
  float wx1 = fx - x0, wy1 = fy - y0;
  float wx0 = 1.0f - wx1, wy0 = 1.0f - wy1;
  int ix0 = (int)x0, iy0 = (int)y0;
  const float* img = yimg + ((long)(b & (K_REFN - 1)) * K_C + c) * (K_H * K_W);
  float a00 = (ix0   >= 0 && ix0   < K_W && iy0   >= 0 && iy0   < K_H) ? img[iy0 * K_W + ix0]           : 0.0f;
  float a10 = (ix0+1 >= 0 && ix0+1 < K_W && iy0   >= 0 && iy0   < K_H) ? img[iy0 * K_W + ix0 + 1]       : 0.0f;
  float a01 = (ix0   >= 0 && ix0   < K_W && iy0+1 >= 0 && iy0+1 < K_H) ? img[(iy0 + 1) * K_W + ix0]     : 0.0f;
  float a11 = (ix0+1 >= 0 && ix0+1 < K_W && iy0+1 >= 0 && iy0+1 < K_H) ? img[(iy0 + 1) * K_W + ix0 + 1] : 0.0f;
  xs[((long)b * K_C + c) * K_NS + n] =
      a00 * wx0 * wy0 + a10 * wx1 * wy0 + a01 * wx0 * wy1 + a11 * wx1 * wy1;
}

// ---------------------------------------------------------------------------
// Stage 9: flash attention.  One wave per (head-batch b', 16-row m tile).
// S tile = one WMMA (K=DH=32); AV consumes 32 keys per WMMA pair.
// grid = (L/16, B*NH), block = 32.
// ---------------------------------------------------------------------------
__global__ void attn_kernel(const float* __restrict__ q_,
                            const float* __restrict__ kfull,
                            const float* __restrict__ vfull,
                            float* __restrict__ attno)
{
  const int lane = threadIdx.x & 31;
  const int lm = lane & 15, kh = lane >> 4;
  const int mt = blockIdx.x;
  const int bp = blockIdx.y;
  const int b = bp / K_NH, h = bp % K_NH;

  __shared__ float ps[16][32];       // P tile (m, n_local) for layout transpose

  // Q fragment: A[m, d] = q_[b, m_global, h*DH + d]
  AFrag qa;
  const float* qbase = q_ + ((long)b * K_L + mt * 16 + lm) * K_C + h * K_DH;
  #pragma unroll
  for (int j = 0; j < 16; ++j) {
    int d = ((j < 8) ? 0 : 16) + kh * 8 + (j & 7);
    qa.h[j] = (_Float16)qbase[d];
  }

  const float* kb = kfull + ((long)b * K_C + h * K_DH) * K_NS;
  const float* vb = vfull + ((long)b * K_C + h * K_DH) * K_NS;

  CFrag o0, o1;
  float rmax[8], rsum[8];
  #pragma unroll
  for (int r = 0; r < 8; ++r) { o0.f[r] = 0.0f; o1.f[r] = 0.0f; rmax[r] = -1e30f; rsum[r] = 0.0f; }

  for (int nc = 0; nc < K_NS; nc += 32) {
    // K fragments: B[d, n] for two 16-wide n sub-tiles
    AFrag kf0, kf1;
    #pragma unroll
    for (int j = 0; j < 16; ++j) {
      int d = ((j < 8) ? 0 : 16) + kh * 8 + (j & 7);
      const float* kr = kb + (long)d * K_NS + nc;
      kf0.h[j] = (_Float16)kr[lm];
      kf1.h[j] = (_Float16)kr[16 + lm];
    }
    CFrag s0, s1;
    #pragma unroll
    for (int r = 0; r < 8; ++r) { s0.f[r] = 0.0f; s1.f[r] = 0.0f; }
    s0.v = wmma16(qa.v, kf0.v, s0.v);
    s1.v = wmma16(qa.v, kf1.v, s1.v);

    // online softmax per row (rows r+8*kh live in this lane's 16-lane group)
    #pragma unroll
    for (int r = 0; r < 8; ++r) {
      float v0 = s0.f[r] * K_SCALE, v1 = s1.f[r] * K_SCALE;
      float tm = fmaxf(v0, v1);
      tm = fmaxf(tm, __shfl_xor(tm, 1, 32));
      tm = fmaxf(tm, __shfl_xor(tm, 2, 32));
      tm = fmaxf(tm, __shfl_xor(tm, 4, 32));
      tm = fmaxf(tm, __shfl_xor(tm, 8, 32));
      float nm = fmaxf(rmax[r], tm);
      float corr = __expf(rmax[r] - nm);
      float p0 = __expf(v0 - nm), p1 = __expf(v1 - nm);
      float tsum = p0 + p1;
      tsum += __shfl_xor(tsum, 1, 32);
      tsum += __shfl_xor(tsum, 2, 32);
      tsum += __shfl_xor(tsum, 4, 32);
      tsum += __shfl_xor(tsum, 8, 32);
      rsum[r] = rsum[r] * corr + tsum;
      rmax[r] = nm;
      o0.f[r] *= corr;
      o1.f[r] *= corr;
      ps[r + 8 * kh][lm]      = p0;   // C layout -> LDS
      ps[r + 8 * kh][16 + lm] = p1;
    }
    __syncthreads();
    // reload P in A-fragment layout
    AFrag pa;
    #pragma unroll
    for (int j = 0; j < 16; ++j) {
      int nl = ((j < 8) ? 0 : 16) + kh * 8 + (j & 7);
      pa.h[j] = (_Float16)ps[lm][nl];
    }
    __syncthreads();

    // V fragments: B[k=n_local, col=d]; n in table order per element
    AFrag vf0, vf1;
    #pragma unroll
    for (int j = 0; j < 16; ++j) {
      int n = nc + ((j < 8) ? 0 : 16) + kh * 8 + (j & 7);
      vf0.h[j] = (_Float16)vb[(long)lm * K_NS + n];
      vf1.h[j] = (_Float16)vb[(long)(16 + lm) * K_NS + n];
    }
    o0.v = wmma16(pa.v, vf0.v, o0.v);
    o1.v = wmma16(pa.v, vf1.v, o1.v);
  }

  // normalize and write attnout[b, m, h*DH + d] (ready for the Wo GEMM)
  #pragma unroll
  for (int r = 0; r < 8; ++r) {
    int m = mt * 16 + r + 8 * kh;
    float inv = 1.0f / rsum[r];
    float* orow = attno + ((long)b * K_L + m) * K_C + h * K_DH;
    orow[lm]      = o0.f[r] * inv;
    orow[16 + lm] = o1.f[r] * inv;
  }
}

// ---------------------------------------------------------------------------
extern "C" void kernel_launch(void* const* d_in, const int* in_sizes, int n_in,
                              void* d_out, int out_size, void* d_ws, size_t ws_size,
                              hipStream_t stream)
{
  (void)in_sizes; (void)n_in; (void)out_size; (void)ws_size;
  const float* x     = (const float*)d_in[0];
  const float* y     = (const float*)d_in[1];
  const float* q_    = (const float*)d_in[2];
  const float* mask_ = (const float*)d_in[3];
  const float* Wq    = (const float*)d_in[4];
  const float* bq    = (const float*)d_in[5];
  const float* Wmd   = (const float*)d_in[6];
  const float* dw_w  = (const float*)d_in[7];
  const float* dw_b  = (const float*)d_in[8];
  const float* ln_g  = (const float*)d_in[9];
  const float* ln_b  = (const float*)d_in[10];
  const float* off_w = (const float*)d_in[11];
  const float* Wk    = (const float*)d_in[12];
  const float* bk    = (const float*)d_in[13];
  const float* Wv    = (const float*)d_in[14];
  const float* bv    = (const float*)d_in[15];
  const float* Wo    = (const float*)d_in[16];
  const float* bo    = (const float*)d_in[17];

  float* ws    = (float*)d_ws;
  float* qm    = ws;                    // 1024
  float* qcond = qm    + 1024;          // 1024
  float* wmod  = qcond + 1024;          // 8*128*128      = 131072
  float* q_t   = wmod  + 131072;        // 8*128*4096     = 4194304
  float* t0    = q_t   + 4194304;       // 8*128*1024     = 1048576
  float* pos   = t0    + 1048576;       // 8*1024*2       = 16384
  float* xs    = pos   + 16384;         // 8*128*1024     = 1048576
  float* kfull = xs    + 1048576;       // 8*128*1024     = 1048576
  float* vfull = kfull + 1048576;       // 8*128*1024     = 1048576
  float* attno = vfull + 1048576;       // 8*2048*128     = 2097152

  float* yo   = (float*)d_out;                        // [8,2048,128]
  float* gout = yo + (size_t)K_B * K_L * K_C;         // [8,1,32,32,2]
  float* rout = gout + (size_t)K_B * K_NS * 2;        // [8,1,32,32,2]

  qmean_kernel<<<dim3(K_B), dim3(K_C), 0, stream>>>(q_, mask_, qm);
  qcond_kernel<<<dim3(K_B), dim3(K_C), 0, stream>>>(qm, Wq, bq, qcond);
  wmod_kernel<<<dim3(K_B * K_C), dim3(K_C), 0, stream>>>(Wmd, qcond, wmod);

  // q_t[b] = wmod[b] @ x[b%4] : M=128 N=4096 K=128
  gemm_wmma_kernel<0, 0, 0><<<dim3(4096 / 16, K_C / 16, K_B), dim3(32), 0, stream>>>(
      wmod, x, nullptr, nullptr, q_t,
      K_C, K_H * K_W, K_H * K_W,
      (long)K_C * K_C, (long)K_C * K_H * K_W, (long)K_C * K_H * K_W,
      K_REFN);

  dwconv_kernel<<<dim3((K_B * K_C * K_NS) / 256), dim3(256), 0, stream>>>(q_t, dw_w, dw_b, t0);
  lngelu_kernel<<<dim3(K_B * K_NS), dim3(K_C), 0, stream>>>(t0, ln_g, ln_b);
  offset_kernel<<<dim3((K_B * K_NS) / 256), dim3(256), 0, stream>>>(t0, off_w, pos, gout, rout);
  gsample_kernel<<<dim3((K_B * K_C * K_NS) / 256), dim3(256), 0, stream>>>(y, pos, xs);

  // K/V projections: M=128 N=1024 K=128, bias per output row
  gemm_wmma_kernel<0, 1, 0><<<dim3(K_NS / 16, K_C / 16, K_B), dim3(32), 0, stream>>>(
      Wk, xs, bk, nullptr, kfull,
      K_C, K_NS, K_NS,
      0L, (long)K_C * K_NS, (long)K_C * K_NS, K_B);
  gemm_wmma_kernel<0, 1, 0><<<dim3(K_NS / 16, K_C / 16, K_B), dim3(32), 0, stream>>>(
      Wv, xs, bv, nullptr, vfull,
      K_C, K_NS, K_NS,
      0L, (long)K_C * K_NS, (long)K_C * K_NS, K_B);

  attn_kernel<<<dim3(K_L / 16, K_B * K_NH), dim3(32), 0, stream>>>(q_, kfull, vfull, attno);

  // yo = attno @ Wo^T + bo : M=B*L=16384 N=128 K=128 (B transposed), bias per col
  gemm_wmma_kernel<1, 0, 1><<<dim3(K_C / 16, (K_B * K_L) / 16, 1), dim3(32), 0, stream>>>(
      attno, Wo, nullptr, bo, yo,
      K_C, K_C, K_C,
      0L, 0L, 0L, 1);
}
